// Attention_20976620274235
// MI455X (gfx1250) — matrix operand order repaired and checked by reading, not compiled, not run on hardware
//
#include <hip/hip_runtime.h>
#include <hip/hip_bf16.h>

// ---------------------------------------------------------------------------
// Attention block for MI455X (gfx1250, wave32, WMMA + TDM).
// Pipeline: cvt(bf16) -> QKV GEMM (wmma bf16) -> RoPE -> flash attention
// (wmma bf16, online softmax, TDM-staged V in LDS) -> output GEMM (f32 out).
// ---------------------------------------------------------------------------

typedef unsigned short ushort_t;
typedef __attribute__((ext_vector_type(16))) __bf16       v16bf;
typedef __attribute__((ext_vector_type(8)))  float        v8f;
typedef __attribute__((ext_vector_type(8)))  unsigned int u32x8;
typedef __attribute__((ext_vector_type(4)))  unsigned int tdm_g0_t;  // uint32x4
typedef __attribute__((ext_vector_type(8)))  int          tdm_g1_t;  // int32x8
typedef __attribute__((ext_vector_type(4)))  int          tdm_g23_t; // int32x4

#define B_    4
#define S_    2048
#define DIM_  2048
#define H_    16
#define HD_   128
#define QKVLD (3 * DIM_)

// LDS tile strides (padded to avoid bank conflicts on strided reads)
#define VLD_  136   // 32 x 136 ushort V chunk (row stride 272B = 17 banks)
#define PLD_  34    // 16 x 34  ushort P tile  (row stride 68B)

// ---- bf16 helpers (storage type = ushort, RNE rounding) -------------------
__device__ __forceinline__ ushort_t f2bf(float f) {
  unsigned int u = __builtin_bit_cast(unsigned int, f);
  u += 0x7FFFu + ((u >> 16) & 1u);
  return (ushort_t)(u >> 16);
}
__device__ __forceinline__ float bf2f(ushort_t h) {
  unsigned int u = ((unsigned int)h) << 16;
  return __builtin_bit_cast(float, u);
}
__device__ __forceinline__ v8f v8f_zero() {
  v8f z;
#pragma unroll
  for (int i = 0; i < 8; ++i) z[i] = 0.f;
  return z;
}

// ---- WMMA fragment loaders (contiguous k order: k = 16*(lane/16) + element, same for A and B) -----
// A fragment, 16x32 bf16, source row-major [16][>=32] at pre-offset tile ptr.
// lane row m = lane&15; K pair = ((lane&16)?16:0) + 2*v
__device__ __forceinline__ v16bf load_frag_a(const ushort_t* tile, int ld, int lane) {
  const ushort_t* p = tile + (size_t)(lane & 15) * ld;
  int kb = (lane & 16) ? 16 : 0;
  u32x8 u;
#pragma unroll
  for (int v = 0; v < 8; ++v) {
    int k = kb + v * 2;
    u[v] = *(const unsigned int*)(p + k);
  }
  return __builtin_bit_cast(v16bf, u);
}

// B fragment, 32x16 bf16, source stored as B^T row-major: src[n][k] (each
// column of B contiguous in k).  lane col n = lane&15;
// K pair = ((lane&16)?16:0) + 2*v
__device__ __forceinline__ v16bf load_frag_bt(const ushort_t* tile, int ld, int lane) {
  const ushort_t* p = tile + (size_t)(lane & 15) * ld;
  int kb = (lane & 16) ? 16 : 0;
  u32x8 u;
#pragma unroll
  for (int v = 0; v < 8; ++v) {
    int k = kb + v * 2;
    u[v] = *(const unsigned int*)(p + k);
  }
  return __builtin_bit_cast(v16bf, u);
}

// B fragment from row-major [keys][d] tile (used on the LDS V chunk).
// element (kk, n) = tile[kk][n]; pairs along kk are ld apart.
__device__ __forceinline__ v16bf load_frag_kn(const ushort_t* tile, int ld, int lane) {
  const ushort_t* p = tile + (lane & 15);
  int kb = (lane & 16) ? 16 : 0;
  u32x8 u;
#pragma unroll
  for (int v = 0; v < 8; ++v) {
    int kk = kb + v * 2;
    unsigned int lo = (unsigned int)p[(size_t)kk * ld];
    unsigned int hi = (unsigned int)p[(size_t)(kk + 1) * ld];
    u[v] = lo | (hi << 16);
  }
  return __builtin_bit_cast(v16bf, u);
}

__device__ __forceinline__ v8f wmma_bf16(v16bf a, v16bf b, v8f c) {
  return __builtin_amdgcn_wmma_f32_16x16x32_bf16(false, a, false, b, (short)0, c,
                                                 false, false);
}

// ---- kernel 1: fp32 -> bf16 convert ---------------------------------------
__global__ void cvt_f32_to_bf16(const float* __restrict__ src,
                                ushort_t* __restrict__ dst, int n) {
  int i = blockIdx.x * blockDim.x + threadIdx.x;
  int stride = gridDim.x * blockDim.x;
  for (; i < n; i += stride) dst[i] = f2bf(src[i]);
}

// ---- kernels 2/5: bf16 WMMA GEMM: C[m][n] = sum_k A[m][k] * Wt[n][k] ------
// One wave (32 threads) computes a 16x64 tile.  F32OUT selects fp32 vs bf16 C.
template <bool F32OUT>
__global__ void gemm_bf16(const ushort_t* __restrict__ A,
                          const ushort_t* __restrict__ Wt,
                          ushort_t* __restrict__ Cb, float* __restrict__ Cf,
                          int K, int lda, int ldw, int ldc) {
  int lane = threadIdx.x;
  int n0 = blockIdx.x * 64;
  int m0 = blockIdx.y * 16;
  v8f acc[4];
#pragma unroll
  for (int j = 0; j < 4; ++j) acc[j] = v8f_zero();

  for (int k0 = 0; k0 < K; k0 += 32) {
    v16bf a = load_frag_a(A + (size_t)m0 * lda + k0, lda, lane);
#pragma unroll
    for (int j = 0; j < 4; ++j) {
      v16bf b = load_frag_bt(Wt + (size_t)(n0 + j * 16) * ldw + k0, ldw, lane);
      acc[j] = wmma_bf16(a, b, acc[j]);
    }
  }
  int col = lane & 15;
  int rb = (lane & 16) ? 8 : 0;
#pragma unroll
  for (int j = 0; j < 4; ++j)
#pragma unroll
    for (int v = 0; v < 8; ++v) {
      size_t idx = (size_t)(m0 + rb + v) * ldc + (size_t)(n0 + j * 16 + col);
      float val = acc[j][v];
      if (F32OUT) Cf[idx] = val; else Cb[idx] = f2bf(val);
    }
}

// ---- kernel 3: RoPE on q,k halves of qkv (in place, bf16) -----------------
__global__ void rope_k(ushort_t* __restrict__ QKV, const float* __restrict__ fc,
                       const float* __restrict__ fs) {
  int idx = blockIdx.x * blockDim.x + threadIdx.x;  // B*S*H*(HD/2) pairs
  int p = idx & 63;
  int hh = (idx >> 6) & (H_ - 1);
  int s = (idx >> 10) & (S_ - 1);
  int b = idx >> 21;
  if (b >= B_) return;
  int d = p * 2;
  float c0 = fc[s * HD_ + d], c1 = fc[s * HD_ + d + 1];
  float s0 = fs[s * HD_ + d], s1 = fs[s * HD_ + d + 1];
  size_t row = ((size_t)b * S_ + s) * QKVLD + (size_t)hh * HD_ + d;
#pragma unroll
  for (int part = 0; part < 2; ++part) {  // q then k
    ushort_t* ptr = QKV + row + part * DIM_;
    float x0 = bf2f(ptr[0]), x1 = bf2f(ptr[1]);
    ptr[0] = f2bf(x0 * c0 - x1 * s0);
    ptr[1] = f2bf(x1 * c1 + x0 * s1);
  }
}

// ---- kernel 4: flash attention, one wave per 16-row q tile per (b,h) ------
// 1 wave/SIMD keeps ~160 live VGPRs register-resident; V chunks are DMA'd
// into LDS by the Tensor Data Mover while the QK^T WMMAs run.
__global__ __attribute__((amdgpu_flat_work_group_size(32, 32),
                          amdgpu_waves_per_eu(1, 1)))
void attn_fa(const ushort_t* __restrict__ QKV,
             const unsigned char* __restrict__ mask,
             ushort_t* __restrict__ Out) {
  int lane = threadIdx.x;
  int q0 = blockIdx.x * 16;
  int h = blockIdx.y;
  int b = blockIdx.z;
  const ushort_t* Qb = QKV + (size_t)b * S_ * QKVLD + (size_t)h * HD_;
  const ushort_t* Kb = Qb + DIM_;
  const ushort_t* Vb = Qb + 2 * DIM_;
  const unsigned char* mk = mask + (size_t)b * S_;

  v16bf aq[4];
#pragma unroll
  for (int dc = 0; dc < 4; ++dc)
    aq[dc] = load_frag_a(Qb + (size_t)q0 * QKVLD + dc * 32, QKVLD, lane);

  v8f acc[8];
#pragma unroll
  for (int n = 0; n < 8; ++n) acc[n] = v8f_zero();
  float mst[8], lst[8];
#pragma unroll
  for (int v = 0; v < 8; ++v) { mst[v] = -1e20f; lst[v] = 0.f; }

  __shared__ ushort_t Vl[32 * VLD_];  // staged V chunk (32 keys x 128 d, padded)
  __shared__ ushort_t Pl[16 * PLD_];  // staged P tile  (16 q x 32 keys)
  const float sm_scale = 0.08838834764831845f;  // 1/sqrt(128)

  // TDM descriptor group1 (invariant): 2-D tile 128 x 32 of 2B elements,
  // tensor_dim0=128, tensor_dim1=2048, dim0_stride=QKVLD, LDS padding of
  // 4 dwords every 64 dwords (row stride 256B -> 272B = VLD_ ushorts).
  // Fields per CDNA5 ISA.
  tdm_g1_t g1;
  g1[0] = (int)((1u << 16)        // data_size = 1 -> 2 bytes
                | (1u << 20)      // pad_enable
                | (5u << 22)      // pad_interval: code 5 = every 64 dwords
                | (3u << 25));    // pad_amount:  code 3 = 4 dwords
  g1[1] = (int)(128u << 16);      // tensor_dim0 = 128 (low 16 in [31:16])
  g1[2] = (int)(2048u << 16);     // tensor_dim1 = 2048 (low 16 in [31:16])
  g1[3] = (int)(128u << 16);      // tile_dim0 = 128
  g1[4] = (int)(32u);             // tile_dim1 = 32, tile_dim2 = 0
  g1[5] = (int)(QKVLD);           // tensor_dim0_stride = 6144 (low 32)
  g1[6] = 0;                      // stride0 hi, stride1 lo
  g1[7] = 0;
  tdm_g23_t gz4;
  gz4[0] = 0; gz4[1] = 0; gz4[2] = 0; gz4[3] = 0;
  tdm_g1_t gz8;
#pragma unroll
  for (int i = 0; i < 8; ++i) gz8[i] = 0;
  unsigned int lds_v = (unsigned int)(unsigned long long)(uintptr_t)(&Vl[0]);

  for (int kc = 0; kc < S_; kc += 32) {
    // ---- TDM: DMA V[kc..kc+32) x 128d into LDS (overlaps QK^T below) ----
    {
      unsigned long long ga =
          (unsigned long long)(uintptr_t)(Vb + (size_t)kc * QKVLD);
      tdm_g0_t g0;
      g0[0] = 1u;                                   // count = 1 valid D#
      g0[1] = lds_v;                                // lds_addr
      g0[2] = (unsigned int)(ga & 0xFFFFFFFFu);     // global_addr[31:0]
      g0[3] = (unsigned int)((ga >> 32) & 0x1FFFFFFu) | (2u << 30);  // type=2
      // amdgpu-toolchain 6-arg form: (g0, g1, g2, g3, g4, cpol)
      __builtin_amdgcn_tensor_load_to_lds(g0, g1, gz4, gz4, gz8, 0);
    }
    // prefetch next chunk's K rows (global_prefetch_b8)
    if (kc + 32 < S_)
      __builtin_prefetch(Kb + (size_t)(kc + 32 + lane) * QKVLD, 0, 0);

    // ---- scores: two 16x16 tiles over the 32-key chunk ----
    v8f s[2];
#pragma unroll
    for (int hh = 0; hh < 2; ++hh) {
      int kb0 = kc + hh * 16;
      v8f t = v8f_zero();
#pragma unroll
      for (int dc = 0; dc < 4; ++dc) {
        v16bf bk = load_frag_bt(Kb + (size_t)kb0 * QKVLD + dc * 32, QKVLD, lane);
        t = wmma_bf16(aq[dc], bk, t);
      }
      bool vis = mk[kb0 + (lane & 15)] != 0;  // column (key) visibility
#pragma unroll
      for (int v = 0; v < 8; ++v)
        t[v] = vis ? t[v] * sm_scale : -3.0e38f;  // masked -> huge negative
      s[hh] = t;
    }
    // ---- online softmax state update (rows live per (v, lane-half)) ----
    float scale_[8];
#pragma unroll
    for (int v = 0; v < 8; ++v) {
      float x = fmaxf(s[0][v], s[1][v]);
#pragma unroll
      for (int off = 1; off < 16; off <<= 1)
        x = fmaxf(x, __shfl_xor(x, off, 32));
      float nm = fmaxf(fmaxf(mst[v], x), -1e20f);  // reference's m clamp
      scale_[v] = __expf(mst[v] - nm);
      mst[v] = nm;
    }
    int rowb = (lane & 16) ? 8 : 0;
    float sums[8];
#pragma unroll
    for (int v = 0; v < 8; ++v) sums[v] = 0.f;
#pragma unroll
    for (int hh = 0; hh < 2; ++hh) {
      int colb = hh * 16 + (lane & 15);
#pragma unroll
      for (int v = 0; v < 8; ++v) {
        float p = __expf(s[hh][v] - mst[v]);  // masked underflows to exactly 0
        Pl[(rowb + v) * PLD_ + colb] = f2bf(p);
        float r = p;
#pragma unroll
        for (int off = 1; off < 16; off <<= 1) r += __shfl_xor(r, off, 32);
        sums[v] += r;
      }
    }
#pragma unroll
    for (int v = 0; v < 8; ++v) lst[v] = lst[v] * scale_[v] + sums[v];
#pragma unroll
    for (int n = 0; n < 8; ++n)
#pragma unroll
      for (int v = 0; v < 8; ++v) acc[n][v] *= scale_[v];

    __builtin_amdgcn_s_wait_tensorcnt(0);  // V chunk landed in LDS
    __syncthreads();                       // Pl stores visible
    v16bf pa = load_frag_a(Pl, PLD_, lane);  // C-layout -> A-layout via LDS
#pragma unroll
    for (int n = 0; n < 8; ++n) {
      v16bf bv = load_frag_kn(Vl + n * 16, VLD_, lane);
      acc[n] = wmma_bf16(pa, bv, acc[n]);
    }
    __syncthreads();  // all Vl/Pl reads done before next TDM/stores
  }
  // epilogue: divide by clamped denom; invisible q rows -> 0 (matches ref)
  int col = lane & 15;
  int rb = (lane & 16) ? 8 : 0;
#pragma unroll
  for (int v = 0; v < 8; ++v) {
    int q = q0 + rb + v;
    float inv = (mk[q] != 0) ? (1.0f / fmaxf(lst[v], 1e-6f)) : 0.0f;
#pragma unroll
    for (int n = 0; n < 8; ++n)
      Out[((size_t)b * S_ + q) * DIM_ + (size_t)h * HD_ + n * 16 + col] =
          f2bf(acc[n][v] * inv);
  }
}

// ---------------------------------------------------------------------------
extern "C" void kernel_launch(void* const* d_in, const int* in_sizes, int n_in,
                              void* d_out, int out_size, void* d_ws,
                              size_t ws_size, hipStream_t stream) {
  (void)in_sizes; (void)n_in; (void)out_size; (void)ws_size;
  const float* x = (const float*)d_in[0];
  const float* fcos = (const float*)d_in[1];
  const float* fsin = (const float*)d_in[2];
  const unsigned char* vmask = (const unsigned char*)d_in[3];  // jnp bool = 1B
  const float* wqkv = (const float*)d_in[4];
  const float* wo = (const float*)d_in[5];
  float* out = (float*)d_out;

  // workspace layout (bf16 buffers), ~201 MB total
  char* ws = (char*)d_ws;
  size_t off = 0;
  ushort_t* x_bf = (ushort_t*)(ws + off);    off += (size_t)B_ * S_ * DIM_ * 2;
  ushort_t* wqkv_bf = (ushort_t*)(ws + off); off += (size_t)QKVLD * DIM_ * 2;
  ushort_t* wo_bf = (ushort_t*)(ws + off);   off += (size_t)DIM_ * DIM_ * 2;
  ushort_t* qkv_bf = (ushort_t*)(ws + off);  off += (size_t)B_ * S_ * QKVLD * 2;
  ushort_t* ao_bf = (ushort_t*)(ws + off);   off += (size_t)B_ * S_ * DIM_ * 2;

  const int n_x = B_ * S_ * DIM_;       // 16,777,216
  const int n_wqkv = QKVLD * DIM_;      // 12,582,912
  const int n_wo = DIM_ * DIM_;         // 4,194,304

  cvt_f32_to_bf16<<<dim3(2048), dim3(256), 0, stream>>>(x, x_bf, n_x);
  cvt_f32_to_bf16<<<dim3(2048), dim3(256), 0, stream>>>(wqkv, wqkv_bf, n_wqkv);
  cvt_f32_to_bf16<<<dim3(1024), dim3(256), 0, stream>>>(wo, wo_bf, n_wo);

  // QKV projection: [8192 x 2048] x [6144 x 2048]^T -> bf16 [8192 x 6144]
  gemm_bf16<false><<<dim3(QKVLD / 64, (B_ * S_) / 16), dim3(32), 0, stream>>>(
      x_bf, wqkv_bf, qkv_bf, nullptr, DIM_, DIM_, DIM_, QKVLD);

  // RoPE on q and k
  rope_k<<<dim3((B_ * S_ * H_ * (HD_ / 2)) / 256), dim3(256), 0, stream>>>(
      qkv_bf, fcos, fsin);

  // flash attention
  attn_fa<<<dim3(S_ / 16, H_, B_), dim3(32), 0, stream>>>(qkv_bf, vmask, ao_bf);

  // output projection: [8192 x 2048] x [2048 x 2048]^T -> f32 d_out
  gemm_bf16<true><<<dim3(DIM_ / 64, (B_ * S_) / 16), dim3(32), 0, stream>>>(
      ao_bf, wo_bf, nullptr, out, DIM_, DIM_, DIM_, DIM_);
}
